// GeneralizedRingAttractorNoGain_24051816857978
// MI455X (gfx1250) — matrix operand order repaired and checked by reading, not compiled, not run on hardware
//
#include <hip/hip_runtime.h>

// ---------- types ----------
typedef __attribute__((ext_vector_type(16))) __bf16 v16bf;
typedef __attribute__((ext_vector_type(8)))  float  v8f;
typedef unsigned int u32x4 __attribute__((ext_vector_type(4)));
typedef int          i32x4 __attribute__((ext_vector_type(4)));
typedef int          i32x8 __attribute__((ext_vector_type(8)));

union Tile {                 // 16 bf16 = 32 bytes = 8 VGPRs
  v16bf v;
  uint4 q[2];
};

#define NN 1024
#define BB 64
#define TT 512
#define NBLK 32
#define BTN (64*512*1024)

__device__ __forceinline__ unsigned short f2bf(float f) {
  unsigned int u = __float_as_uint(f);
  u += 0x7FFFu + ((u >> 16) & 1u);          // round to nearest even
  return (unsigned short)(u >> 16);
}

// ---------- grid-wide barrier (agent scope) ----------
__device__ __forceinline__ void grid_sync(unsigned int* cnt, unsigned int* gen) {
  __syncthreads();
  if (threadIdx.x == 0) {
    unsigned int g = __hip_atomic_load(gen, __ATOMIC_RELAXED, __HIP_MEMORY_SCOPE_AGENT);
    unsigned int prev = __hip_atomic_fetch_add(cnt, 1u, __ATOMIC_ACQ_REL, __HIP_MEMORY_SCOPE_AGENT);
    if (prev == (unsigned)NBLK - 1u) {
      __hip_atomic_store(cnt, 0u, __ATOMIC_RELAXED, __HIP_MEMORY_SCOPE_AGENT);
      __hip_atomic_fetch_add(gen, 1u, __ATOMIC_RELEASE, __HIP_MEMORY_SCOPE_AGENT);
    } else {
      while (__hip_atomic_load(gen, __ATOMIC_ACQUIRE, __HIP_MEMORY_SCOPE_AGENT) == g) {
        __builtin_amdgcn_s_sleep(2);
      }
    }
  }
  __syncthreads();
}

// ---------- TDM: DMA a contiguous 32KB bf16 A-tile (16 rows x 1024) into LDS ----------
// D# group0: count=1, lds_addr, 57-bit global_addr, type=2.
// D# group1: data_size=2B (code 1), tensor_dim0=tile_dim0=16384 elems, tile_dim1=1,
//            tensor_dim0_stride=16384, workgroup_mask=0 (not in a cluster).
__device__ __forceinline__ void tdm_load_tile(unsigned int lds_byte_addr,
                                              const unsigned short* gsrc) {
  unsigned long long ga = (unsigned long long)(const void*)gsrc;
  u32x4 g0 = { 1u,                                   // count=1, user descriptor
               lds_byte_addr,                        // lds_addr
               (unsigned int)(ga & 0xFFFFFFFFu),     // global_addr[31:0]
               (unsigned int)((ga >> 32) & 0x01FFFFFFu) | 0x80000000u }; // addr[56:32] | type=2
  i32x8 g1 = { (int)(1u << 16),                      // data_size = 2 bytes
               (int)0x40000000,                      // tensor_dim0[15:0]=16384 in high half
               (int)0x00010000,                      // tensor_dim0 hi=0 | tensor_dim1=1
               (int)0x40000000,                      // tensor_dim1 hi=0 | tile_dim0=16384
               1,                                    // tile_dim1=1, tile_dim2=0
               16384,                                // tensor_dim0_stride[31:0]
               0, 0 };
  i32x4 z4 = { 0, 0, 0, 0 };
#if defined(__clang_major__) && (__clang_major__ >= 23)
  i32x8 z8 = { 0, 0, 0, 0, 0, 0, 0, 0 };
  __builtin_amdgcn_tensor_load_to_lds(g0, g1, z4, z4, z8, 0);
#else
  __builtin_amdgcn_tensor_load_to_lds(g0, g1, z4, z4, 0);
#endif
}

// ---------- prep: bf16 weight panels + r init + barrier init ----------
__global__ void ring_prep(const float* __restrict__ Wo, const float* __restrict__ Wa,
                          const float* __restrict__ r0,
                          unsigned short* __restrict__ Wb, float* __restrict__ rf,
                          unsigned short* __restrict__ rb, unsigned int* __restrict__ bar) {
  int i = blockIdx.x * blockDim.x + threadIdx.x;
  if (i < 2) bar[i] = 0u;
  if (i < BB * NN) {                       // parity-0 r buffer used at t=0
    float v = r0[i];
    rf[i] = v;
    rb[i] = f2bf(v);
  }
  if (i < 3 * NN * NN) {
    int mat  = i >> 20;
    int rem  = i & 0xFFFFF;
    int kk   = rem & 31;
    int cidx = rem >> 5;                   // kt*1024 + ncol
    int ncol = cidx & 1023;
    int kt   = cidx >> 10;
    int k    = kt * 32 + kk;
    float w;
    if (mat == 0) w = 0.1f * Wo[ncol * NN + k] - 0.1f;     // J1*Wo + J0
    else          w = Wa[(mat - 1) * NN * NN + ncol * NN + k];
    Wb[i] = f2bf(w);
  }
}

// ---------- persistent time-loop kernel ----------
__global__ void __launch_bounds__(256, 1)
ring_run(const float* __restrict__ act, float* __restrict__ out,
         const unsigned short* __restrict__ Wb, float* __restrict__ rf,
         unsigned short* __restrict__ rb, unsigned int* __restrict__ bar) {
  __shared__ __align__(128) unsigned short ldsA[16 * NN];   // 32 KB shared A tile
  __shared__ float redC[256];
  __shared__ float redS[256];

  const int tid  = threadIdx.x;
  const int lane = tid & 31;
  const int wave = tid >> 5;
  const int tileId = blockIdx.x * 8 + wave;   // 0..255
  const int mt = tileId >> 6;                 // 0..3   (batch tile; uniform per block)
  const int nt = tileId & 63;                 // 0..63  (neuron tile)
  const int l15 = lane & 15;
  const int hi  = lane >> 4;

  // scalar-only copies for the TDM descriptor (wave 0 issues it)
  const int mt0 = (blockIdx.x * 8) >> 6;
  const unsigned int ldsA_addr = (unsigned int)(unsigned long long)(void*)&ldsA[0];

  // A operand (16-bit 16x32 layout): lane row = l15 within the LDS tile;
  // elements 0..15 = K base..base+15, base = hi?16:0
  const unsigned short* ldsAp = ldsA + l15 * NN + (hi ? 16 : 0);
  // B operand: column = nt*16+l15, K chunk = hi*16 within the 32-contiguous-K panel
  const int ncol = nt * 16 + l15;
  const int boff = (ncol << 5) + (hi << 4);
  // C/D layout: VGPR j holds M = j + 8*hi
  const int rbase = mt * 16 + (hi << 3);

  // phase-B ids: 2 batch rows per block, 128 threads each
  const int half = tid >> 7;
  const int t2   = tid & 127;
  const int bbat = blockIdx.x * 2 + half;

  float cosv[8], sinv[8];
#pragma unroll
  for (int j = 0; j < 8; ++j) {
    float th = 6.283185307179586f * (float)(t2 + j * 128) * (1.0f / 1024.0f);
    __sincosf(th, &sinv[j], &cosv[j]);
  }

  // stage A tile for t=0 (reads parity-0 bf16 r buffer)
  if (wave == 0) {
    tdm_load_tile(ldsA_addr, rb + mt0 * 16 * NN);
    __builtin_amdgcn_s_wait_tensorcnt(0);
  }
  __syncthreads();

  for (int t = 0; t < TT; ++t) {
    const int rd = t & 1, wr = rd ^ 1;

    // ---- phase A: rec = baseW@r + a0*Wa0@r + a1*Wa1@r (A from LDS, B from L2) ----
    v8f c0 = {}, c1 = {}, c2 = {};
#pragma unroll 2
    for (int kt = 0; kt < 32; ++kt) {
      Tile a, b0, b1, b2;
      const unsigned short* ap = ldsAp + kt * 32;
      a.q[0] = *(const uint4*)(ap);
      a.q[1] = *(const uint4*)(ap + 8);
      const unsigned short* bp = Wb + (kt << 15) + boff;   // kt*1024*32
      b0.q[0] = *(const uint4*)(bp);
      b0.q[1] = *(const uint4*)(bp + 8);
      b1.q[0] = *(const uint4*)(bp + (1 << 20));
      b1.q[1] = *(const uint4*)(bp + (1 << 20) + 8);
      b2.q[0] = *(const uint4*)(bp + (2 << 20));
      b2.q[1] = *(const uint4*)(bp + (2 << 20) + 8);
      if (kt + 1 < 32) __builtin_prefetch(bp + (1 << 15), 0, 3);   // next K panel (L2-resident)
      c0 = __builtin_amdgcn_wmma_f32_16x16x32_bf16(false, a.v, false, b0.v, (short)0, c0, false, false);
      c1 = __builtin_amdgcn_wmma_f32_16x16x32_bf16(false, a.v, false, b1.v, (short)0, c1, false, false);
      c2 = __builtin_amdgcn_wmma_f32_16x16x32_bf16(false, a.v, false, b2.v, (short)0, c2, false, false);
    }

    // combine with per-batch action scalars, tanh, leaky update, write bump history
#pragma unroll
    for (int j = 0; j < 8; ++j) {
      const int row = rbase + j;
      const float a0 = act[row * (TT * 2) + t * 2 + 0];
      const float a1 = act[row * (TT * 2) + t * 2 + 1];
      const float rec = c0[j] + a0 * c1[j] + a1 * c2[j];
      const float th  = tanhf(rec);
      const int idx = row * NN + ncol;
      const float rn = 0.85f * rf[rd * (BB * NN) + idx] + 0.15f * th;
      rf[wr * (BB * NN) + idx] = rn;
      rb[wr * (BB * NN) + idx] = f2bf(rn);
      out[BTN + (row * TT + t) * NN + ncol] = rn;          // bump_hist (B,T,N)
    }

    grid_sync(bar, bar + 1);

    // issue next step's A-tile DMA now: overlaps with phase B, waited below
    if ((t + 1 < TT) && wave == 0) {
      tdm_load_tile(ldsA_addr, rb + wr * (BB * NN) + mt0 * 16 * NN);
    }

    // ---- phase B: rd7 via rank-2 decomposition cos(2pi(i-j)/N) = ci*cj + si*sj ----
    const float* rrow = rf + wr * (BB * NN) + bbat * NN;
    float cs = 0.f, ss = 0.f;
#pragma unroll
    for (int j = 0; j < 8; ++j) {
      const float rv = rrow[t2 + j * 128];
      cs += cosv[j] * rv;
      ss += sinv[j] * rv;
    }
    redC[tid] = cs; redS[tid] = ss;
    __syncthreads();
    for (int ofs = 64; ofs > 0; ofs >>= 1) {
      if (t2 < ofs) { redC[tid] += redC[tid + ofs]; redS[tid] += redS[tid + ofs]; }
      __syncthreads();
    }
    const float Cb = redC[half << 7], Sb = redS[half << 7];
    __syncthreads();

    float v[8], mx = -3.4e38f;
#pragma unroll
    for (int j = 0; j < 8; ++j) {
      v[j] = Cb * cosv[j] + Sb * sinv[j];
      mx = fmaxf(mx, v[j]);
    }
    redC[tid] = mx;
    __syncthreads();
    for (int ofs = 64; ofs > 0; ofs >>= 1) {
      if (t2 < ofs) redC[tid] = fmaxf(redC[tid], redC[tid + ofs]);
      __syncthreads();
    }
    const float inv = 1.0f / redC[half << 7];
#pragma unroll
    for (int j = 0; j < 8; ++j)
      out[(bbat * TT + t) * NN + t2 + j * 128] = v[j] * inv;   // rd7 history (B,T,N)

    // complete the next-step A tile before anyone reads it
    if (wave == 0) __builtin_amdgcn_s_wait_tensorcnt(0);
    __syncthreads();
  }
}

extern "C" void kernel_launch(void* const* d_in, const int* in_sizes, int n_in,
                              void* d_out, int out_size, void* d_ws, size_t ws_size,
                              hipStream_t stream) {
  const float* act = (const float*)d_in[0];   // (64,512,2)
  const float* r0  = (const float*)d_in[1];   // (64,1024)
  const float* Wo  = (const float*)d_in[2];   // (1024,1024)
  const float* Wa  = (const float*)d_in[3];   // (2,1024,1024)
  float* out = (float*)d_out;

  char* ws = (char*)d_ws;
  unsigned int*   bar = (unsigned int*)(ws);                      // 2 uints
  unsigned short* Wb  = (unsigned short*)(ws + 256);              // 6 MB bf16 panels
  float*          rf  = (float*)(ws + 256 + 6291456);             // 2 x 256 KB fp32 r
  unsigned short* rb  = (unsigned short*)(ws + 256 + 6291456 + 524288); // 2 x 128 KB bf16 r

  ring_prep<<<12288, 256, 0, stream>>>(Wo, Wa, r0, Wb, rf, rb, bar);
  ring_run<<<NBLK, 256, 0, stream>>>(act, out, Wb, rf, rb, bar);
}
